// SelfAttention_4758823764273
// MI455X (gfx1250) — matrix operand order repaired and checked by reading, not compiled, not run on hardware
//
#include <hip/hip_runtime.h>
#include <hip/hip_bf16.h>

// ---------------------------------------------------------------------------
// Self-attention for MI455X (gfx1250, wave32, WMMA 16x16x32 f16 -> f32 acc).
//   B=2, S=4096, E=512, H=8, D=64.
//   (0) W^T -> f16
//   (1) QKV projection via WMMA; Q pre-scaled by 0.125*log2(e); V stored
//       transposed [b][h][d][s].
//   (2) flash attention, transposed form: S^T = K Q^T (softmax per-lane),
//       O^T = V^T P^T.  K/V chunks staged in LDS (shared by 8 waves/block),
//       double-buffered via async global->LDS DMA.
// ---------------------------------------------------------------------------

typedef __attribute__((ext_vector_type(16))) _Float16 v16h;
typedef __attribute__((ext_vector_type(8)))  _Float16 v8h;
typedef __attribute__((ext_vector_type(8)))  float    v8f;
typedef int v4i __attribute__((vector_size(16)));     // matches builtin param

#define WMMA_F16(a, b, c) \
  __builtin_amdgcn_wmma_f32_16x16x32_f16(false, (a), false, (b), (short)0, (c), false, false)

#define SEQ   4096
#define EMB   512
#define NH    8
#define HD    64
// fold softmax scale 1/sqrt(64)=0.125 and log2(e) into Q so we can use exp2
#define QSCALE (0.125f * 1.44269504088896340736f)

// Padded LDS row strides (halves) -> conflict-free 16-lane fragment reads
#define KROW  72   // 32 key rows x 64 d (+8 pad)
#define VROW  40   // 64 d rows x 32 s (+8 pad)

#if __has_builtin(__builtin_amdgcn_global_load_async_to_lds_b128) && \
    __has_builtin(__builtin_amdgcn_s_wait_asynccnt)
#define USE_ASYNC_LDS 1
#define GLOBAL_AS __attribute__((address_space(1)))
#define LDS_AS    __attribute__((address_space(3)))
// g, l are generic pointers; route through AS-qualified void* (known-good cast),
// then to the exact v4i32 pointee the builtin expects.
#define ASYNC_CP_B128(g, l)                                                   \
  __builtin_amdgcn_global_load_async_to_lds_b128(                             \
      (GLOBAL_AS v4i*)(GLOBAL_AS void*)(g),                                   \
      (LDS_AS    v4i*)(LDS_AS    void*)(l), 0, 0)
#else
#define USE_ASYNC_LDS 0
#endif

// A-fragment (16x32, f16 source): lane row handled by caller (rowPtr),
// element e -> K = kbase + lg*16 + e  => two contiguous 8-half chunks.
__device__ inline v16h load_afrag_f16(const _Float16* __restrict__ rowPtr, int kbase, int lg) {
  v8h c0 = *(const v8h*)(rowPtr + kbase + lg * 16);
  v8h c1 = *(const v8h*)(rowPtr + kbase + 8 + lg * 16);
  v16h a;
#pragma unroll
  for (int i = 0; i < 8; ++i) { a[i] = c0[i]; a[8 + i] = c1[i]; }
  return a;
}

// A-fragment from fp32 source (convert on the fly).
__device__ inline v16h load_afrag_f32(const float* __restrict__ rowPtr, int kbase, int lg) {
  const float* p0 = rowPtr + kbase + lg * 16;
  const float* p1 = rowPtr + kbase + 8 + lg * 16;
  float4 a0 = *(const float4*)p0;
  float4 a1 = *(const float4*)(p0 + 4);
  float4 b0 = *(const float4*)p1;
  float4 b1 = *(const float4*)(p1 + 4);
  v16h r;
  r[0]  = (_Float16)a0.x; r[1]  = (_Float16)a0.y; r[2]  = (_Float16)a0.z; r[3]  = (_Float16)a0.w;
  r[4]  = (_Float16)a1.x; r[5]  = (_Float16)a1.y; r[6]  = (_Float16)a1.z; r[7]  = (_Float16)a1.w;
  r[8]  = (_Float16)b0.x; r[9]  = (_Float16)b0.y; r[10] = (_Float16)b0.z; r[11] = (_Float16)b0.w;
  r[12] = (_Float16)b1.x; r[13] = (_Float16)b1.y; r[14] = (_Float16)b1.z; r[15] = (_Float16)b1.w;
  return r;
}

// ---------------------------------------------------------------------------
// Kernel 0: W^T (n-major) in f16 so WMMA B-fragments are one 32B load.
// ---------------------------------------------------------------------------
__global__ __launch_bounds__(256) void prep_wt(const float* __restrict__ Wq,
                                               const float* __restrict__ Wk,
                                               const float* __restrict__ Wv,
                                               _Float16* __restrict__ wt) {
  const int z = blockIdx.y;
  const float* W = (z == 0) ? Wq : ((z == 1) ? Wk : Wv);
  const int idx = blockIdx.x * 256 + threadIdx.x;   // 0..262143
  const int n = idx >> 9;
  const int k = idx & 511;
  wt[(size_t)z * EMB * EMB + idx] = (_Float16)W[(size_t)k * EMB + n];
}

// ---------------------------------------------------------------------------
// Kernel 1: QKV projection.  Wave: 16 rows x 64 cols, 64 WMMAs.
// ---------------------------------------------------------------------------
__global__ __launch_bounds__(256) void qkv_proj(const float*    __restrict__ x,
                                                const _Float16* __restrict__ wt,
                                                const float*    __restrict__ bq,
                                                const float*    __restrict__ bk,
                                                const float*    __restrict__ bv,
                                                _Float16* __restrict__ qf,
                                                _Float16* __restrict__ kf,
                                                _Float16* __restrict__ vt) {
  const int wave = threadIdx.x >> 5;
  const int lane = threadIdx.x & 31;
  const int ln   = lane & 15;
  const int lg   = lane >> 4;

  const int rowTile = blockIdx.x * 8 + wave;   // 0..511
  const int h       = blockIdx.y;
  const int proj    = blockIdx.z;              // 0=Q 1=K 2=V

  const int row0 = rowTile * 16;
  const float* xrow = x + (size_t)(row0 + ln) * EMB;
  const _Float16* wbase = wt + (size_t)proj * EMB * EMB + (size_t)h * HD * EMB;

  v8f acc[4];
#pragma unroll
  for (int t = 0; t < 4; ++t) acc[t] = (v8f){};

  for (int ks = 0; ks < EMB; ks += 32) {
    v16h a = load_afrag_f32(xrow, ks, lg);
#pragma unroll
    for (int t = 0; t < 4; ++t) {
      v16h b = *(const v16h*)(wbase + (size_t)(t * 16 + ln) * EMB + ks + lg * 16);
      acc[t] = WMMA_F16(a, b, acc[t]);
    }
  }

  const float* bias = (proj == 0) ? bq : ((proj == 1) ? bk : bv);
  const int b_ = row0 >> 12;
  const int s0 = row0 & (SEQ - 1);
  const size_t bh = (size_t)b_ * NH + h;

#pragma unroll
  for (int t = 0; t < 4; ++t) {
    const int d  = t * 16 + ln;
    const float bv_ = bias[h * HD + d];
    if (proj == 0) {
      _Float16* dst = qf + (bh * SEQ) * HD + d;
#pragma unroll
      for (int r = 0; r < 8; ++r)
        dst[(size_t)(s0 + r + 8 * lg) * HD] = (_Float16)((acc[t][r] + bv_) * QSCALE);
    } else if (proj == 1) {
      _Float16* dst = kf + (bh * SEQ) * HD + d;
#pragma unroll
      for (int r = 0; r < 8; ++r)
        dst[(size_t)(s0 + r + 8 * lg) * HD] = (_Float16)(acc[t][r] + bv_);
    } else {
      v8h pack;
#pragma unroll
      for (int r = 0; r < 8; ++r) pack[r] = (_Float16)(acc[t][r] + bv_);
      *(v8h*)(vt + (bh * HD + d) * SEQ + s0 + 8 * lg) = pack;
    }
  }
}

// ---------------------------------------------------------------------------
// Kernel 2: flash attention.  Block = 8 waves sharing one (b,h); each wave
// owns one 16-query tile.  Per 32-key chunk: cooperative stage of K (32x64)
// and V^T (64x32) into padded LDS (double buffered, async DMA), then per wave
// 4 WMMAs for S^T, online softmax, 4 WMMAs for O^T.
// ---------------------------------------------------------------------------
__global__ __launch_bounds__(256) void flash_attn(const _Float16* __restrict__ qf,
                                                  const _Float16* __restrict__ kf,
                                                  const _Float16* __restrict__ vt,
                                                  float* __restrict__ out) {
  __shared__ _Float16 kbuf[2][32 * KROW];
  __shared__ _Float16 vbuf[2][64 * VROW];

  const int tid  = threadIdx.x;
  const int wave = tid >> 5;
  const int lane = tid & 31;
  const int ln   = lane & 15;
  const int lg   = lane >> 4;

  const int bh = blockIdx.x >> 5;                      // 0..15, shared by block
  const int qt = (blockIdx.x & 31) * 8 + wave;         // query tile 0..255
  const int b_ = bh >> 3;
  const int h  = bh & 7;
  const int q0 = qt * 16;

  const _Float16* qbase = qf + (size_t)bh * SEQ * HD;
  const _Float16* kbase = kf + (size_t)bh * SEQ * HD;
  const _Float16* vbase = vt + (size_t)bh * HD * SEQ;

  // per-thread cooperative copy slots (one b128 each for K and V per chunk)
  const int krow = tid >> 3, kseg = tid & 7;           // 32 rows x 8 segs
  const int vrow = tid >> 2, vseg = tid & 3;           // 64 rows x 4 segs

  // Q B-fragments (fixed per wave): column n = query q0+ln, e -> d = lg*16+e (+32)
  const _Float16* qrow = qbase + (size_t)(q0 + ln) * HD;
  const v16h qb0 = *(const v16h*)(qrow + lg * 16);
  const v16h qb1 = *(const v16h*)(qrow + 32 + lg * 16);

  v8f ot0 = (v8f){}, ot1 = (v8f){}, ot2 = (v8f){}, ot3 = (v8f){};
  float mrow = -3.0e38f, lrow = 0.0f;

  // ---- stage chunk 0 into buffer 0 ----------------------------------------
  {
    const _Float16* kg = kbase + (size_t)krow * HD + kseg * 8;
    const _Float16* vg = vbase + (size_t)vrow * SEQ + vseg * 8;
    _Float16* kl = &kbuf[0][krow * KROW + kseg * 8];
    _Float16* vl = &vbuf[0][vrow * VROW + vseg * 8];
#if USE_ASYNC_LDS
    ASYNC_CP_B128(kg, kl);
    ASYNC_CP_B128(vg, vl);
    __builtin_amdgcn_s_wait_asynccnt(0);
#else
    *(v8h*)kl = *(const v8h*)kg;
    *(v8h*)vl = *(const v8h*)vg;
#endif
  }
  __syncthreads();

  for (int jc = 0; jc < SEQ / 32; ++jc) {
    const int cur = jc & 1;
    const int nxt = cur ^ 1;

    // ---- kick off DMA for the next chunk (overlaps with WMMA below) -------
    if (jc + 1 < SEQ / 32) {
      const int j1 = (jc + 1) * 32;
      const _Float16* kg = kbase + (size_t)(j1 + krow) * HD + kseg * 8;
      const _Float16* vg = vbase + (size_t)vrow * SEQ + j1 + vseg * 8;
      _Float16* kl = &kbuf[nxt][krow * KROW + kseg * 8];
      _Float16* vl = &vbuf[nxt][vrow * VROW + vseg * 8];
#if USE_ASYNC_LDS
      ASYNC_CP_B128(kg, kl);
      ASYNC_CP_B128(vg, vl);
#else
      *(v8h*)kl = *(const v8h*)kg;
      *(v8h*)vl = *(const v8h*)vg;
#endif
    }

    // ---- S^T tiles from LDS: rows = keys (local 0..31), cols = queries ----
    const _Float16* krow0 = &kbuf[cur][(0  + ln) * KROW];
    const _Float16* krow1 = &kbuf[cur][(16 + ln) * KROW];
    v8f st0, st1;
    {
      v16h ka = load_afrag_f16(krow0, 0, lg);
      st0 = WMMA_F16(ka, qb0, (v8f){});
      ka = load_afrag_f16(krow0, 32, lg);
      st0 = WMMA_F16(ka, qb1, st0);
      ka = load_afrag_f16(krow1, 0, lg);
      st1 = WMMA_F16(ka, qb0, (v8f){});
      ka = load_afrag_f16(krow1, 32, lg);
      st1 = WMMA_F16(ka, qb1, st1);
    }

    // ---- online softmax (lane = query column) -----------------------------
    float cmax = st0[0];
#pragma unroll
    for (int r = 0; r < 8; ++r) { cmax = fmaxf(cmax, st0[r]); cmax = fmaxf(cmax, st1[r]); }
    cmax = fmaxf(cmax, __shfl_xor(cmax, 16, 32));
    const float mnew  = fmaxf(mrow, cmax);
    const float alpha = __builtin_amdgcn_exp2f(mrow - mnew);

    float csum = 0.0f;
#pragma unroll
    for (int r = 0; r < 8; ++r) {
      float p0 = __builtin_amdgcn_exp2f(st0[r] - mnew);
      float p1 = __builtin_amdgcn_exp2f(st1[r] - mnew);
      st0[r] = p0; st1[r] = p1;
      csum += p0 + p1;
    }
    csum += __shfl_xor(csum, 16, 32);
    lrow = lrow * alpha + csum;
    mrow = mnew;

#pragma unroll
    for (int r = 0; r < 8; ++r) {
      ot0[r] *= alpha; ot1[r] *= alpha; ot2[r] *= alpha; ot3[r] *= alpha;
    }

    // ---- P^T B-fragment: lane column = query, element e -> key lg*16+e ----
    float sh0[8], sh1[8];
#pragma unroll
    for (int r = 0; r < 8; ++r) {
      sh0[r] = __shfl_xor(st0[r], 16, 32);
      sh1[r] = __shfl_xor(st1[r], 16, 32);
    }
    v16h pb;
#pragma unroll
    for (int e = 0; e < 8; ++e) {
      pb[e]     = (_Float16)((lg == 0) ? st0[e] : sh1[e]);
      pb[8 + e] = (_Float16)((lg == 0) ? sh0[e] : st1[e]);
    }

    // ---- O^T += V^T P^T (V^T rows = d from LDS, contraction = 32 keys) ----
    ot0 = WMMA_F16(load_afrag_f16(&vbuf[cur][(0 * 16 + ln) * VROW], 0, lg), pb, ot0);
    ot1 = WMMA_F16(load_afrag_f16(&vbuf[cur][(1 * 16 + ln) * VROW], 0, lg), pb, ot1);
    ot2 = WMMA_F16(load_afrag_f16(&vbuf[cur][(2 * 16 + ln) * VROW], 0, lg), pb, ot2);
    ot3 = WMMA_F16(load_afrag_f16(&vbuf[cur][(3 * 16 + ln) * VROW], 0, lg), pb, ot3);

#if USE_ASYNC_LDS
    __builtin_amdgcn_s_wait_asynccnt(0);
#endif
    __syncthreads();
  }

  // ---- epilogue: out[b][q0+ln][h*64 + d], d = t*16 + 8*lg + r -------------
  const float inv = 1.0f / lrow;
  float* orow = out + ((size_t)b_ * SEQ + q0 + ln) * EMB + h * HD;
  const v8f* ots[4] = { &ot0, &ot1, &ot2, &ot3 };
#pragma unroll
  for (int t = 0; t < 4; ++t) {
    const v8f o = *ots[t];
    float4 lo = { o[0] * inv, o[1] * inv, o[2] * inv, o[3] * inv };
    float4 hi = { o[4] * inv, o[5] * inv, o[6] * inv, o[7] * inv };
    float* p = orow + t * 16 + 8 * lg;
    *(float4*)(p)     = lo;
    *(float4*)(p + 4) = hi;
  }
}

// ---------------------------------------------------------------------------
// Launch
// ---------------------------------------------------------------------------
extern "C" void kernel_launch(void* const* d_in, const int* in_sizes, int n_in,
                              void* d_out, int out_size, void* d_ws, size_t ws_size,
                              hipStream_t stream) {
  const float* x  = (const float*)d_in[0];
  const float* Wq = (const float*)d_in[1];
  const float* bq = (const float*)d_in[2];
  const float* Wk = (const float*)d_in[3];
  const float* bk = (const float*)d_in[4];
  const float* Wv = (const float*)d_in[5];
  const float* bv = (const float*)d_in[6];
  float* out = (float*)d_out;

  char* ws = (char*)d_ws;
  const size_t WT_BYTES  = (size_t)3 * EMB * EMB * sizeof(_Float16);   // 1.5 MB
  const size_t QKV_BYTES = (size_t)2 * SEQ * EMB * sizeof(_Float16);   // 8 MB each
  _Float16* wt = (_Float16*)ws;
  _Float16* qf = (_Float16*)(ws + WT_BYTES);
  _Float16* kf = (_Float16*)(ws + WT_BYTES + QKV_BYTES);
  _Float16* vt = (_Float16*)(ws + WT_BYTES + 2 * QKV_BYTES);

  prep_wt<<<dim3((EMB * EMB) / 256, 3), 256, 0, stream>>>(Wq, Wk, Wv, wt);
  qkv_proj<<<dim3(64, NH, 3), 256, 0, stream>>>(x, wt, bq, bk, bv, qf, kf, vt);
  flash_attn<<<dim3(512), 256, 0, stream>>>(qf, kf, vt, out);
}
